// SimpleImage2Text_74964359185005
// MI455X (gfx1250) — matrix operand order repaired and checked by reading, not compiled, not run on hardware
//
#include <hip/hip_runtime.h>
#include <cstdint>
#include <cstddef>

// ---------------------------------------------------------------------------
// SimpleImage2Text teacher-forcing forward for MI455X (gfx1250, wave32).
//
//   k_build_x   : gather decoder inputs, split f32 -> bf16 hi/lo
//   k_split_mat : split W_ih / W_hh / W_fc f32 -> bf16 hi/lo
//   k_gemm      : C = A @ B^T + bias, bf16x3 (hi*hi + hi*lo + lo*hi) via
//                 v_wmma_f32_16x16x32_bf16. A/B chunk tiles are DMA'd into
//                 double-buffered LDS by the Tensor Data Mover
//                 (tensor_load_to_lds, TENSORcnt, LDS row padding for
//                 conflict-free ds_load_b128 fragment reads).
//   k_gru_scan  : 1 persistent WGP, 8 waves; per-step gh = h @ W_hh^T with
//                 bf16x3 WMMA (W_hh hi/lo streamed from L2, h hi/lo in LDS),
//                 f32 gates, s_barrier sync, prefetch of next-step gx.
//   k_gemm      : logits = hs @ W_fc^T + b_fc (M=4096,N=32000,K=256),
//                 the dominant 67-GFLOP compute-bound phase.
// ---------------------------------------------------------------------------

#define HDIM 256
#define GDIM 768      // 3*H
#define TLEN 512
#define BATCH 8
#define MTOT 4096     // BATCH*TLEN
#define VOCAB 32000

#define GEMM_AROWS 128
#define GEMM_BROWS 64
#define LROW 40       // LDS row pitch in ushorts: 64B data + 16B TDM pad = 80B
#define HS 264        // GRU h hi/lo LDS row pitch (ushorts): 528B, 16B aligned

typedef __attribute__((ext_vector_type(16))) __bf16 v16bf;
typedef __attribute__((ext_vector_type(8)))  __bf16 v8bf;
typedef __attribute__((ext_vector_type(8)))  float  v8f;
typedef unsigned int v4u __attribute__((ext_vector_type(4)));
typedef int v8i __attribute__((ext_vector_type(8)));
typedef int v4i __attribute__((ext_vector_type(4)));

__device__ __forceinline__ unsigned short f32_to_bf16_rne(float x) {
  union { float f; unsigned u; } a; a.f = x;
  unsigned r = a.u + 0x7fffu + ((a.u >> 16) & 1u);
  return (unsigned short)(r >> 16);
}
__device__ __forceinline__ float bf16_bits_to_f32(unsigned short h) {
  union { unsigned u; float f; } a; a.u = ((unsigned)h) << 16;
  return a.f;
}
__device__ __forceinline__ v16bf load_bf16x16(const unsigned short* p0,
                                              const unsigned short* p1) {
  v8bf a = *(const v8bf*)p0;
  v8bf b = *(const v8bf*)p1;
  return __builtin_shufflevector(a, b, 0,1,2,3,4,5,6,7,8,9,10,11,12,13,14,15);
}
__device__ __forceinline__ v8f v8f_zero() {
  v8f z = {0.f,0.f,0.f,0.f,0.f,0.f,0.f,0.f};
  return z;
}
__device__ __forceinline__ unsigned lds_offset(const void* p) {
  return (unsigned)(unsigned long long)p;   // low 32 bits = LDS byte offset
}

// --- TDM: DMA a [tile_rows x 32] bf16 tile (row pitch 256 elems) into LDS --
// D# per CDNA5 ISA: 2D tensor, data_size=2B,
// pad_enable, pad_interval=16 DW (one 64B row), pad_amount=4 DW (16B)
// -> LDS row pitch 80B, keeping ds_load_b128 fragment reads 16B-aligned.
__device__ __forceinline__ void tdm_load_tile(const unsigned short* src,
                                              unsigned tile_rows,
                                              unsigned lds_byte) {
  unsigned long long ga = (unsigned long long)src;
  v4u g0;
  g0[0] = 1u;                                    // count=1 (valid), user mode
  g0[1] = lds_byte;                              // lds_addr
  g0[2] = (unsigned)ga;                          // global_addr[31:0]
  g0[3] = ((unsigned)(ga >> 32) & 0x01FFFFFFu) | 0x80000000u; // addr[56:32]|type=2
  v8i g1;
  g1[0] = (int)((1u << 16)      // data_size = 1 (2 bytes)
              | (1u << 20)      // pad_enable
              | (3u << 22)      // pad_interval = 16 DWORDs
              | (3u << 25));    // pad_amount   = 4 DWORDs (16B)
  g1[1] = (int)(32u << 16);                        // tensor_dim0 = 32 (lo16)
  g1[2] = (int)((tile_rows & 0xFFFFu) << 16);      // dim0 hi | tensor_dim1 lo
  g1[3] = (int)((tile_rows >> 16) | (32u << 16));  // dim1 hi | tile_dim0 = 32
  g1[4] = (int)(tile_rows & 0xFFFFu);              // tile_dim1 | tile_dim2=0
  g1[5] = (int)HDIM;                               // tensor_dim0_stride = 256
  g1[6] = 0;
  g1[7] = 0;
  v4i gz4 = {0, 0, 0, 0};
  v8i gz8 = {0, 0, 0, 0, 0, 0, 0, 0};
  // 6-arg toolchain form: (g0, g1, g2, g3, g4, cpol)
  __builtin_amdgcn_tensor_load_to_lds(g0, g1, gz4, gz4, gz8, 0);
}

// --- build decoder-input matrix x[4096][256], split into bf16 hi/lo --------
__global__ void __launch_bounds__(256) k_build_x(
    const float* __restrict__ features, const int* __restrict__ targets,
    const float* __restrict__ emb,
    unsigned short* __restrict__ x_hi, unsigned short* __restrict__ x_lo) {
  const int m = blockIdx.x;           // m = b*TLEN + t
  const int b = m >> 9;
  const int t = m & (TLEN - 1);
  const int j = threadIdx.x;
  float v;
  if (t == 0) {
    v = features[b * HDIM + j];
  } else {
    const int tok = targets[b * TLEN + (t - 1)];
    v = emb[(size_t)tok * HDIM + j];
  }
  const unsigned short hi = f32_to_bf16_rne(v);
  x_hi[(size_t)m * HDIM + j] = hi;
  x_lo[(size_t)m * HDIM + j] = f32_to_bf16_rne(v - bf16_bits_to_f32(hi));
}

// --- split a [rows x 256] f32 matrix into bf16 hi/lo -----------------------
__global__ void __launch_bounds__(256) k_split_mat(
    const float* __restrict__ W,
    unsigned short* __restrict__ hi, unsigned short* __restrict__ lo) {
  const size_t idx = (size_t)blockIdx.x * HDIM + threadIdx.x;
  const float v = W[idx];
  const unsigned short h = f32_to_bf16_rne(v);
  hi[idx] = h;
  lo[idx] = f32_to_bf16_rne(v - bf16_bits_to_f32(h));
}

// --- bf16x3 WMMA GEMM with TDM-fed double-buffered LDS ---------------------
// block = 256 thr = 8 waves; wave tile 16x64; wg tile 128M x 64N; K = 256.
__global__ void __launch_bounds__(256) k_gemm(
    const unsigned short* __restrict__ Ah, const unsigned short* __restrict__ Al,
    const unsigned short* __restrict__ Bh, const unsigned short* __restrict__ Bl,
    const float* __restrict__ bias, float* __restrict__ C, int N) {
  __shared__ unsigned short sA[2][2][GEMM_AROWS * LROW];  // [stage][hi/lo]
  __shared__ unsigned short sB[2][2][GEMM_BROWS * LROW];

  const int lane = threadIdx.x & 31;
  const int wave = threadIdx.x >> 5;
  const int m0 = blockIdx.y * GEMM_AROWS;
  const int n0 = blockIdx.x * GEMM_BROWS;

  auto issue_chunk = [&](int kc8) {
    const size_t koff = (size_t)kc8 * 32;
    const int st = kc8 & 1;
    tdm_load_tile(Ah + (size_t)m0 * HDIM + koff, GEMM_AROWS, lds_offset(&sA[st][0][0]));
    tdm_load_tile(Al + (size_t)m0 * HDIM + koff, GEMM_AROWS, lds_offset(&sA[st][1][0]));
    tdm_load_tile(Bh + (size_t)n0 * HDIM + koff, GEMM_BROWS, lds_offset(&sB[st][0][0]));
    tdm_load_tile(Bl + (size_t)n0 * HDIM + koff, GEMM_BROWS, lds_offset(&sB[st][1][0]));
  };

  if (wave == 0) { issue_chunk(0); issue_chunk(1); }

  v8f acc[4];
#pragma unroll
  for (int nt = 0; nt < 4; ++nt) acc[nt] = v8f_zero();

  const int arow = wave * 16 + (lane & 15);
  const int akb = (lane >> 4) << 3;   // A half-wave K offset: 0 / 8
  const int bkb = (lane >> 4) << 3;   // B half-wave K offset: 0 / 8

#pragma unroll 1
  for (int kc8 = 0; kc8 < 8; ++kc8) {
    const int st = kc8 & 1;
    if (wave == 0) {
      if (kc8 == 7) __builtin_amdgcn_s_wait_tensorcnt(0);
      else          __builtin_amdgcn_s_wait_tensorcnt(4);
    }
    __syncthreads();   // chunk kc8 resident in LDS for all waves

    const unsigned short* ah = &sA[st][0][arow * LROW + akb];
    const unsigned short* al = &sA[st][1][arow * LROW + akb];
    const v16bf a_hi = load_bf16x16(ah, ah + 16);
    const v16bf a_lo = load_bf16x16(al, al + 16);
#pragma unroll
    for (int nt = 0; nt < 4; ++nt) {
      const int brow = nt * 16 + (lane & 15);
      const unsigned short* bh = &sB[st][0][brow * LROW + bkb];
      const unsigned short* bl = &sB[st][1][brow * LROW + bkb];
      const v16bf b_hi = load_bf16x16(bh, bh + 16);
      const v16bf b_lo = load_bf16x16(bl, bl + 16);
      acc[nt] = __builtin_amdgcn_wmma_f32_16x16x32_bf16(
          false, a_hi, false, b_hi, (short)0, acc[nt], false, false);
      acc[nt] = __builtin_amdgcn_wmma_f32_16x16x32_bf16(
          false, a_hi, false, b_lo, (short)0, acc[nt], false, false);
      acc[nt] = __builtin_amdgcn_wmma_f32_16x16x32_bf16(
          false, a_lo, false, b_hi, (short)0, acc[nt], false, false);
    }
    __syncthreads();   // all waves done reading stage st
    if (wave == 0 && kc8 < 6) issue_chunk(kc8 + 2);
  }

#pragma unroll
  for (int nt = 0; nt < 4; ++nt) {
    const int n = n0 + nt * 16 + (lane & 15);
    const float bv = bias[n];
#pragma unroll
    for (int i = 0; i < 8; ++i) {
      const int m = m0 + wave * 16 + i + ((lane >> 4) << 3);
      C[(size_t)m * N + n] = acc[nt][i] + bv;
    }
  }
}

// --- GRU scan: 1 workgroup, 8 waves; bf16x3 WMMA recurrent GEMM ------------
__global__ void __launch_bounds__(256) k_gru_scan(
    const float* __restrict__ gx,            // [MTOT][768]
    const unsigned short* __restrict__ WhhH, // [768][256] bf16 hi
    const unsigned short* __restrict__ WhhL, // [768][256] bf16 lo
    const float* __restrict__ bhh,           // [768]
    unsigned short* __restrict__ hs_hi,      // [MTOT][256] bf16 hi
    unsigned short* __restrict__ hs_lo)      // [MTOT][256] bf16 lo
{
  __shared__ float hbuf[BATCH * HDIM];          // f32 state (exact)
  __shared__ float ghbuf[BATCH * GDIM];         // gate staging
  __shared__ unsigned short hbf_hi[16 * HS];    // h hi, M padded 8->16 w/ 0
  __shared__ unsigned short hbf_lo[16 * HS];    // h lo

  const int tid  = threadIdx.x;
  const int lane = tid & 31;
  const int wave = tid >> 5;

  for (int i = tid; i < BATCH * HDIM; i += 256) hbuf[i] = 0.0f;
  for (int i = tid; i < 16 * HS; i += 256) { hbf_hi[i] = 0; hbf_lo[i] = 0; }
  __syncthreads();

  const int am  = lane & 15;          // padded batch row
  const int akb = (lane >> 4) << 3;
  const int bkb = (lane >> 4) << 3;

  for (int t = 0; t < TLEN; ++t) {
    // ---- phase 1: gh = h @ Whh^T for this wave's 96 gate columns ----
    v8f acc[6];
#pragma unroll
    for (int nt = 0; nt < 6; ++nt) acc[nt] = v8f_zero();

#pragma unroll 1
    for (int kc8 = 0; kc8 < 8; ++kc8) {
      const int k0 = kc8 * 32;
      const unsigned short* hp = &hbf_hi[am * HS + k0 + akb];
      const unsigned short* lp = &hbf_lo[am * HS + k0 + akb];
      const v16bf a_hi = load_bf16x16(hp, hp + 16);
      const v16bf a_lo = load_bf16x16(lp, lp + 16);
#pragma unroll
      for (int nt = 0; nt < 6; ++nt) {
        const int n = wave * 96 + nt * 16 + (lane & 15);
        const unsigned short* bh = WhhH + (size_t)n * HDIM + k0 + bkb;
        const unsigned short* bl = WhhL + (size_t)n * HDIM + k0 + bkb;
        const v16bf b_hi = load_bf16x16(bh, bh + 16);
        const v16bf b_lo = load_bf16x16(bl, bl + 16);
        acc[nt] = __builtin_amdgcn_wmma_f32_16x16x32_bf16(
            false, a_hi, false, b_hi, (short)0, acc[nt], false, false);
        acc[nt] = __builtin_amdgcn_wmma_f32_16x16x32_bf16(
            false, a_hi, false, b_lo, (short)0, acc[nt], false, false);
        acc[nt] = __builtin_amdgcn_wmma_f32_16x16x32_bf16(
            false, a_lo, false, b_hi, (short)0, acc[nt], false, false);
      }
    }
    if (lane < 16) {   // lanes 0-15 hold batch rows 0-7 (vgpr i <-> row i)
#pragma unroll
      for (int nt = 0; nt < 6; ++nt) {
        const int n = wave * 96 + nt * 16 + lane;
#pragma unroll
        for (int i = 0; i < 8; ++i) ghbuf[i * GDIM + n] = acc[nt][i];
      }
    }
    __syncthreads();

    // ---- phase 2: gates + state update; thread owns column j = tid ----
    const int j = tid;
#pragma unroll
    for (int b = 0; b < BATCH; ++b) {
      const size_t m = (size_t)b * TLEN + t;
      const float* gxr = gx + m * GDIM;
      const float gr = ghbuf[b * GDIM + j]            + bhh[j];
      const float gz = ghbuf[b * GDIM + j + HDIM]     + bhh[j + HDIM];
      const float gn = ghbuf[b * GDIM + j + 2 * HDIM] + bhh[j + 2 * HDIM];
      const float r  = 1.0f / (1.0f + __expf(-(gxr[j] + gr)));
      const float z  = 1.0f / (1.0f + __expf(-(gxr[j + HDIM] + gz)));
      const float nn = tanhf(gxr[j + 2 * HDIM] + r * gn);
      const float hp = hbuf[b * HDIM + j];
      const float hn = (1.0f - z) * nn + z * hp;
      hbuf[b * HDIM + j] = hn;
      const unsigned short hi = f32_to_bf16_rne(hn);
      const unsigned short lo = f32_to_bf16_rne(hn - bf16_bits_to_f32(hi));
      hbf_hi[b * HS + j] = hi;
      hbf_lo[b * HS + j] = lo;
      hs_hi[m * HDIM + j] = hi;
      hs_lo[m * HDIM + j] = lo;
      if (t + 1 < TLEN) __builtin_prefetch(gxr + GDIM, 0, 1);  // next-step gx
    }
    __syncthreads();
  }
}

// ---------------------------------------------------------------------------
extern "C" void kernel_launch(void* const* d_in, const int* in_sizes, int n_in,
                              void* d_out, int out_size, void* d_ws, size_t ws_size,
                              hipStream_t stream) {
  (void)in_sizes; (void)n_in; (void)out_size; (void)ws_size;
  const float* features = (const float*)d_in[0];
  const int*   targets  = (const int*)  d_in[1];
  const float* emb      = (const float*)d_in[2];
  const float* W_ih     = (const float*)d_in[3];
  const float* b_ih     = (const float*)d_in[4];
  const float* W_hh     = (const float*)d_in[5];
  const float* b_hh     = (const float*)d_in[6];
  const float* W_fc     = (const float*)d_in[7];
  const float* b_fc     = (const float*)d_in[8];
  float* out = (float*)d_out;

  char* ws = (char*)d_ws;
  size_t off = 0;
  auto take = [&](size_t bytes) -> char* {
    char* p = ws + off;
    off += (bytes + 255) & ~(size_t)255;
    return p;
  };
  unsigned short* x_hi   = (unsigned short*)take((size_t)MTOT  * HDIM * 2);
  unsigned short* x_lo   = (unsigned short*)take((size_t)MTOT  * HDIM * 2);
  unsigned short* wih_hi = (unsigned short*)take((size_t)GDIM  * HDIM * 2);
  unsigned short* wih_lo = (unsigned short*)take((size_t)GDIM  * HDIM * 2);
  unsigned short* whh_hi = (unsigned short*)take((size_t)GDIM  * HDIM * 2);
  unsigned short* whh_lo = (unsigned short*)take((size_t)GDIM  * HDIM * 2);
  unsigned short* wfc_hi = (unsigned short*)take((size_t)VOCAB * HDIM * 2);
  unsigned short* wfc_lo = (unsigned short*)take((size_t)VOCAB * HDIM * 2);
  unsigned short* hs_hi  = (unsigned short*)take((size_t)MTOT  * HDIM * 2);
  unsigned short* hs_lo  = (unsigned short*)take((size_t)MTOT  * HDIM * 2);
  float*          gx     = (float*)         take((size_t)MTOT  * GDIM * 4);

  // 1. inputs
  k_build_x<<<MTOT, 256, 0, stream>>>(features, targets, emb, x_hi, x_lo);
  k_split_mat<<<GDIM, 256, 0, stream>>>(W_ih, wih_hi, wih_lo);
  k_split_mat<<<GDIM, 256, 0, stream>>>(W_hh, whh_hi, whh_lo);
  k_split_mat<<<VOCAB, 256, 0, stream>>>(W_fc, wfc_hi, wfc_lo);

  // 2. gx = x @ W_ih^T + b_ih   (M=4096, N=768)
  k_gemm<<<dim3(GDIM / 64, MTOT / 128), 256, 0, stream>>>(
      x_hi, x_lo, wih_hi, wih_lo, b_ih, gx, GDIM);

  // 3. sequential GRU scan (latency-bound; single persistent WGP)
  k_gru_scan<<<1, 256, 0, stream>>>(gx, whh_hi, whh_lo, b_hh, hs_hi, hs_lo);

  // 4. logits = hs @ W_fc^T + b_fc  (M=4096, N=32000) — dominant 67 GFLOP
  k_gemm<<<dim3(VOCAB / 64, MTOT / 128), 256, 0, stream>>>(
      hs_hi, hs_lo, wfc_hi, wfc_lo, b_fc, out, VOCAB);
}
